// CompGNN_45037027066322
// MI455X (gfx1250) — matrix operand order repaired and checked by reading, not compiled, not run on hardware
//
#include <hip/hip_runtime.h>
#include <hip/hip_bf16.h>

// Problem constants (fixed by the reference).
#define NE_   100000      // num_ent
#define NR_   500         // num_rel
#define NN_   100500      // num_nodes
#define MN_   100512      // num_nodes padded to 16
#define MR_   512         // num_rel padded to 16
#define EDG_  600000      // total edges
#define EA_   200000      // end of opt_e group
#define EB_   400000      // end of ipt_e group (rel edge groups provably dead)

typedef __attribute__((ext_vector_type(16))) _Float16     v16h;
typedef __attribute__((ext_vector_type(8)))  float        v8f;
typedef __attribute__((ext_vector_type(4)))  unsigned int u32x4;
typedef __attribute__((ext_vector_type(4)))  float        f32x4;

union FragH { v16h v; u32x4 q[2]; };

// ---------------------------------------------------------------- utilities
__global__ void fill_f32_k(float* __restrict__ p, float v, long long n) {
  long long i = (long long)blockIdx.x * blockDim.x + threadIdx.x;
  long long s = (long long)gridDim.x * blockDim.x;
  for (; i < n; i += s) p[i] = v;
}

__global__ void cvt_f32_f16_k(const float* __restrict__ src,
                              _Float16* __restrict__ dst, long long n) {
  long long i = (long long)blockIdx.x * blockDim.x + threadIdx.x;
  long long s = (long long)gridDim.x * blockDim.x;
  for (; i < n; i += s) dst[i] = (_Float16)src[i];
}

// h[:,0:256] = x  (h row stride 768)
__global__ void copy_x_to_h_k(const float* __restrict__ x, float* __restrict__ h) {
  long long n = (long long)NN_ * 256;
  long long i = (long long)blockIdx.x * blockDim.x + threadIdx.x;
  long long s = (long long)gridDim.x * blockDim.x;
  for (; i < n; i += s) {
    long long r = i >> 8;
    int c = (int)(i & 255);
    h[r * 768 + c] = x[i];
  }
}

// -------------------------------------------------------- degrees and norms
__global__ void count_deg_k(const int* __restrict__ ei,
                            float* __restrict__ dr0, float* __restrict__ dc0,
                            float* __restrict__ dr1, float* __restrict__ dc1) {
  int e = blockIdx.x * blockDim.x + threadIdx.x;
  if (e >= EB_) return;
  int r = ei[e];
  int c = ei[EDG_ + e];
  float* dr = (e < EA_) ? dr0 : dr1;
  float* dc = (e < EA_) ? dc0 : dc1;
  __hip_atomic_fetch_add(dr + r, 1.0f, __ATOMIC_RELAXED, __HIP_MEMORY_SCOPE_AGENT);
  __hip_atomic_fetch_add(dc + c, 1.0f, __ATOMIC_RELAXED, __HIP_MEMORY_SCOPE_AGENT);
}

__global__ void compute_nrm_k(const int* __restrict__ ei,
                              const float* __restrict__ dr0, const float* __restrict__ dc0,
                              const float* __restrict__ dr1, const float* __restrict__ dc1,
                              float* __restrict__ nrm) {
  int e = blockIdx.x * blockDim.x + threadIdx.x;
  if (e >= EB_) return;
  int r = ei[e];
  int c = ei[EDG_ + e];
  const float* dr = (e < EA_) ? dr0 : dr1;
  const float* dc = (e < EA_) ? dc0 : dc1;
  float a = dr[r], b = dc[c];
  float ia = (a > 0.0f) ? (1.0f / sqrtf(a)) : 0.0f;
  float ib = (b > 0.0f) ? (1.0f / sqrtf(b)) : 0.0f;
  nrm[e] = ia * ib;
}

// ------------------------------------------------------------------ scatter
// cat[row, coloff:coloff+256] += xW[col, :] * nrm[e];  one edge per block, 64 lanes x float4
__global__ void scatter_k(const int* __restrict__ ei, const float* __restrict__ nrm,
                          const float* __restrict__ xW, float* __restrict__ cat,
                          int elo, int coloff) {
  int e = elo + blockIdx.x;
  int row = ei[e];
  int col = ei[EDG_ + e];
  float s = nrm[e];
  const f32x4* src = (const f32x4*)(xW + (long long)col * 256);
  float* dst = cat + (long long)row * 768 + coloff + threadIdx.x * 4;
  f32x4 v = src[threadIdx.x];
  __hip_atomic_fetch_add(dst + 0, v.x * s, __ATOMIC_RELAXED, __HIP_MEMORY_SCOPE_AGENT);
  __hip_atomic_fetch_add(dst + 1, v.y * s, __ATOMIC_RELAXED, __HIP_MEMORY_SCOPE_AGENT);
  __hip_atomic_fetch_add(dst + 2, v.z * s, __ATOMIC_RELAXED, __HIP_MEMORY_SCOPE_AGENT);
  __hip_atomic_fetch_add(dst + 3, v.w * s, __ATOMIC_RELAXED, __HIP_MEMORY_SCOPE_AGENT);
}

// ------------------------------------------------------------------- GEMM
// C[M,N] = A[M,K](f16,row-major) @ B[K,N](f16,row-major), f32 accumulate via
// v_wmma_f32_16x16x32_f16. One wave per 16x16 output tile; 8 waves/block share
// the N-tile (B rows hit L0/L2). Optional bias + double-tanh epilogue, optional
// f16 mirror store (next-layer activations). Stores bounded by Mstore.
__global__ void gemm_wmma_f16_k(const _Float16* __restrict__ A, int lda,
                                const _Float16* __restrict__ B, int ldb,
                                float* __restrict__ C, int ldc,
                                _Float16* __restrict__ C16, int ldc16,
                                const float* __restrict__ bias,
                                int K, int Mpad, int Mstore, int act) {
  const int lane = threadIdx.x & 31;
  const int wv   = threadIdx.x >> 5;
  const int m0   = (blockIdx.y * 8 + wv) * 16;
  if (m0 >= Mpad) return;                      // uniform per wave: EXEC stays full
  const int n0   = blockIdx.x * 16;

  // A fragment: lane -> row m0+(lane&15); halves [ka,ka+8) and [ka+16,ka+24)
  const int ka = (lane >> 4) << 3;
  const _Float16* ap = A + (long long)(m0 + (lane & 15)) * lda + ka;
  // B fragment: lane -> col n0+(lane&15); K-rows [ka,ka+8) and [ka+16,ka+24)
  const _Float16* bp = B + (long long)ka * ldb + n0 + (lane & 15);

  v8f acc = {};
  for (int kk = 0; kk < K; kk += 32) {
    FragH a, b;
    a.q[0] = *(const u32x4*)(ap);
    a.q[1] = *(const u32x4*)(ap + 16);
#pragma unroll
    for (int s = 0; s < 8; ++s) {
      b.v[s]     = bp[(long long)s * ldb];
      b.v[8 + s] = bp[(long long)(16 + s) * ldb];
    }
    acc = __builtin_amdgcn_wmma_f32_16x16x32_f16(
        false, a.v, false, b.v, (short)0, acc, false, false);
    ap += 32;
    bp += (long long)32 * ldb;
  }

  const int n  = n0 + (lane & 15);
  const int mb = m0 + ((lane >> 4) << 3);
  const float bv = bias ? bias[n] : 0.0f;
#pragma unroll
  for (int j = 0; j < 8; ++j) {
    int m = mb + j;
    if (m >= Mstore) continue;
    float v = acc[j] + bv;
    if (act) v = tanhf(tanhf(v));              // tanh in _conv, tanh again in driver
    C[(long long)m * ldc + n] = v;
    if (C16) C16[(long long)m * ldc16 + n] = (_Float16)v;
  }
}

// ------------------------------------------------------------------- host
struct LayerP {
  const float *w_opt_e, *w_loop_e, *w_ipt_e, *w_loop_r;
  const float *ent_w, *ent_b, *rel_w, *rel_b;
};

extern "C" void kernel_launch(void* const* d_in, const int* in_sizes, int n_in,
                              void* d_out, int out_size, void* d_ws, size_t ws_size,
                              hipStream_t stream) {
  // ---- locate inputs by size; map params under either flattening order ----
  const float* x = nullptr;
  const int* ei = nullptr;
  const float* P[32];
  int np = 0;
  for (int i = 0; i < n_in; ++i) {
    long long s = in_sizes[i];
    if (s == (long long)NN_ * 256 && !x) { x = (const float*)d_in[i]; continue; }
    if (s == 2LL * EDG_ && !ei)          { ei = (const int*)d_in[i];   continue; }
    if (s == 1) continue;                 // num_ent / num_rel scalars (known)
    if (np < 32) P[np++] = (const float*)d_in[i];
  }
  LayerP Lp[2];
  const float *FW = nullptr, *FB = nullptr;
  bool insertion = (np >= 1 && in_sizes[0] >= 0) && true;
  // detect: insertion order starts with a 256x256 weight (65536 elems)
  // sorted-key order starts with final_b (256) then final_w (196608)
  long long p0 = 0;
  { // re-scan to fetch the flat size of P[0]
    int seen = 0;
    for (int i = 0; i < n_in; ++i) {
      long long s = in_sizes[i];
      if (s == (long long)NN_ * 256 || s == 2LL * EDG_ || s == 1) continue;
      if (seen == 0) { p0 = s; }
      ++seen; if (seen) break;
    }
  }
  insertion = (p0 == 65536);
  if (insertion) {
    for (int l = 0; l < 2; ++l) {
      int b = l * 10;
      Lp[l].w_opt_e  = P[b + 0]; Lp[l].w_loop_e = P[b + 1];
      Lp[l].w_ipt_e  = P[b + 2]; /* w_opt_r P[b+3] dead */
      Lp[l].w_loop_r = P[b + 4]; /* w_ipt_r P[b+5] dead */
      Lp[l].ent_w = P[b + 6]; Lp[l].ent_b = P[b + 7];
      Lp[l].rel_w = P[b + 8]; Lp[l].rel_b = P[b + 9];
    }
    FW = P[20]; FB = P[21];
  } else { // sorted-by-key pytree order
    FB = P[0]; FW = P[1];
    for (int l = 0; l < 2; ++l) {
      int b = 2 + l * 10;
      Lp[l].ent_b = P[b + 0]; Lp[l].ent_w = P[b + 1];
      Lp[l].rel_b = P[b + 2]; Lp[l].rel_w = P[b + 3];
      Lp[l].w_ipt_e  = P[b + 4]; /* w_ipt_r P[b+5] dead */
      Lp[l].w_loop_e = P[b + 6]; Lp[l].w_loop_r = P[b + 7];
      Lp[l].w_opt_e  = P[b + 8]; /* w_opt_r P[b+9] dead */
    }
  }

  // ---- workspace arena -----------------------------------------------------
  char* w = (char*)d_ws;
  auto arena = [&](size_t bytes) { void* p = w; w += (bytes + 255) & ~(size_t)255; return p; };
  _Float16* X16a  = (_Float16*)arena((size_t)MN_ * 256 * 2);
  _Float16* X16b  = (_Float16*)arena((size_t)MN_ * 256 * 2);
  float*    H     = (float*)   arena((size_t)MN_ * 768 * 4);
  float*    CAT   = (float*)   arena((size_t)NE_ * 768 * 4);   // reused as H16 at the end
  _Float16* CAT16 = (_Float16*)arena((size_t)NE_ * 768 * 2);
  float*    XW    = (float*)   arena((size_t)NE_ * 256 * 4);
  _Float16* W16   = (_Float16*)arena((size_t)768 * 256 * 2);
  float*    RT    = (float*)   arena((size_t)MR_ * 256 * 4);
  _Float16* RT16  = (_Float16*)arena((size_t)MR_ * 256 * 2);
  float*    DEG   = (float*)   arena((size_t)4 * MN_ * 4);
  float*    NRM   = (float*)   arena((size_t)EB_ * 4);
  float *DR0 = DEG, *DC0 = DEG + MN_, *DR1 = DEG + 2 * MN_, *DC1 = DEG + 3 * MN_;

  auto fill = [&](float* p, float v, long long n) {
    int blocks = (int)((n + 255) / 256); if (blocks > 4096) blocks = 4096;
    fill_f32_k<<<blocks, 256, 0, stream>>>(p, v, n);
  };
  auto cvt = [&](const float* s, _Float16* d, long long n) {
    int blocks = (int)((n + 255) / 256); if (blocks > 4096) blocks = 4096;
    cvt_f32_f16_k<<<blocks, 256, 0, stream>>>(s, d, n);
  };
  auto gemm = [&](const _Float16* A, int lda, const _Float16* B, int ldb,
                  float* C, int ldc, _Float16* C16, int ldc16, const float* bias,
                  int N, int K, int Mpad, int Mstore, int act) {
    dim3 g(N / 16, (Mpad / 16 + 7) / 8);
    gemm_wmma_f16_k<<<g, 256, 0, stream>>>(A, lda, B, ldb, C, ldc, C16, ldc16,
                                           bias, K, Mpad, Mstore, act);
  };

  // ---- stage 0: init h, activations, degrees, norms ------------------------
  fill(H, 0.0f, (long long)MN_ * 768);
  copy_x_to_h_k<<<4096, 256, 0, stream>>>(x, H);
  cvt(x, X16a, (long long)NN_ * 256);
  // zero pad rows of both activation buffers (12 rows x 256 halves = 1536 f32)
  fill((float*)(X16a + (long long)NN_ * 256), 0.0f, (long long)(MN_ - NN_) * 128);
  fill((float*)(X16b + (long long)NN_ * 256), 0.0f, (long long)(MN_ - NN_) * 128);
  fill(DEG, 0.0f, 4LL * MN_);
  count_deg_k<<<(EB_ + 255) / 256, 256, 0, stream>>>(ei, DR0, DC0, DR1, DC1);
  compute_nrm_k<<<(EB_ + 255) / 256, 256, 0, stream>>>(ei, DR0, DC0, DR1, DC1, NRM);

  // ---- stage 1: two CompGCN layers ----------------------------------------
  for (int L = 0; L < 2; ++L) {
    const LayerP& p = Lp[L];
    _Float16* cur = (L == 0) ? X16a : X16b;
    _Float16* nxt = (L == 0) ? X16b : X16a;
    int off = (L + 1) * 256;

    fill(CAT, 0.0f, (long long)NE_ * 768);
    // loop_e -> cat[:,256:512] (direct store into zeroed slice)
    cvt(p.w_loop_e, W16, 256 * 256);
    gemm(cur, 256, W16, 256, CAT + 256, 768, nullptr, 0, nullptr, 256, 256, NE_, NE_, 0);
    // opt_e: xW = x @ W, then scatter-add into cat[:,0:256]
    cvt(p.w_opt_e, W16, 256 * 256);
    gemm(cur, 256, W16, 256, XW, 256, nullptr, 0, nullptr, 256, 256, NE_, NE_, 0);
    scatter_k<<<EA_, 64, 0, stream>>>(ei, NRM, XW, CAT, 0, 0);
    // ipt_e: into cat[:,512:768]
    cvt(p.w_ipt_e, W16, 256 * 256);
    gemm(cur, 256, W16, 256, XW, 256, nullptr, 0, nullptr, 256, 256, NE_, NE_, 0);
    scatter_k<<<EB_ - EA_, 64, 0, stream>>>(ei, NRM, XW, CAT, EA_, 512);
    // entity output: tanh(tanh(cat @ ent_w + ent_b)) -> h[:,off] and nxt(f16)
    cvt(CAT, CAT16, (long long)NE_ * 768);
    cvt(p.ent_w, W16, 768 * 256);
    gemm(CAT16, 768, W16, 256, H + off, 768, nxt, 256, p.ent_b, 256, 768, NE_, NE_, 1);
    // relation path (rel-edge aggregations provably contribute nothing):
    // r = tanh(tanh( (x_rel @ w_loop_r) @ rel_w[256:512] + rel_b ))
    cvt(p.w_loop_r, W16, 256 * 256);
    gemm(cur + (long long)NE_ * 256, 256, W16, 256, RT, 256, nullptr, 0, nullptr,
         256, 256, MR_, MR_, 0);
    cvt(RT, RT16, (long long)MR_ * 256);
    cvt(p.rel_w, W16, 768 * 256);
    gemm(RT16, 256, W16 + 256 * 256, 256, H + (long long)NE_ * 768 + off, 768,
         nxt + (long long)NE_ * 256, 256, p.rel_b, 256, 256, MR_, NR_, 1);
  }

  // ---- stage 2: final projection  out = h @ final_w + final_b -------------
  _Float16* H16 = (_Float16*)CAT;  // reuse arena (cat no longer needed)
  cvt(H, H16, (long long)MN_ * 768);
  cvt(FW, W16, 768 * 256);
  gemm(H16, 768, W16, 256, (float*)d_out, 256, nullptr, 0, FB, 256, 768, MN_, NN_, 0);
  (void)out_size; (void)ws_size;
}
